// TGN_3178275799043
// MI455X (gfx1250) — matrix operand order repaired and checked by reading, not compiled, not run on hardware
//
#include <hip/hip_runtime.h>
#include <hip/hip_bf16.h>

typedef __attribute__((ext_vector_type(16))) _Float16 v16h;
typedef __attribute__((ext_vector_type(8)))  float    v8f;

#define N_NODES_  200000
#define N_EDGES_  1000000
#define BATCH_    2000
#define KNBR_     20
#define D_        128
#define DQ_       256
#define DK_       384
#define RAW_      512
#define DMSG_     100
#define MROWS_    (3*BATCH_)     // 6000
#define IDROWS_   (2*BATCH_)     // 4000

__device__ __forceinline__ float sigmoidf_(float x) { return 1.0f / (1.0f + expf(-x)); }

// ---------------------------------------------------------------------------
// Repack Wk/Wv (DK x DQ, f32 row-major) into f16 WMMA B-fragment order.
// Fragment (kk, nt): halves laid out [lane][e], where
//   B[e] = W[kk*32 + (lane>=16?16:0) + e][nt*16 + (lane&15)]
// ---------------------------------------------------------------------------
__global__ void repack_kernel(const float* __restrict__ Wk, const float* __restrict__ Wv,
                              _Float16* __restrict__ Wk16, _Float16* __restrict__ Wv16) {
    const int per = (DK_/32)*(DQ_/16)*512;   // 12*16*512 = 98304
    int idx = blockIdx.x * 256 + threadIdx.x;
    if (idx >= 2*per) return;
    const float* W = (idx < per) ? Wk : Wv;
    _Float16*    O = (idx < per) ? Wk16 : Wv16;
    int t    = idx % per;
    int e    = t & 15;
    int lane = (t >> 4) & 31;
    int frag = t >> 9;
    int nt   = frag & 15;
    int kk   = frag >> 4;
    int row  = kk*32 + ((lane >= 16) ? 16 : 0) + e;
    int col  = nt*16 + (lane & 15);
    O[t] = (_Float16)W[row*DQ_ + col];
}

// ---------------------------------------------------------------------------
// Fused attention + embedding MLP. One block per query row m (0..5999).
// ---------------------------------------------------------------------------
__global__ __launch_bounds__(256) void attn_emb_kernel(
    const float* __restrict__ node_feats, const float* __restrict__ memory,
    const float* __restrict__ edge_feats, const float* __restrict__ edge_times,
    const float* __restrict__ neigh_times,
    const int* __restrict__ src_ids, const int* __restrict__ dst_ids, const int* __restrict__ neg_ids,
    const int* __restrict__ neigh_ids, const int* __restrict__ neigh_eidx,
    const float* __restrict__ time_w, const float* __restrict__ time_b,
    const float* __restrict__ Wq, const float* __restrict__ bq,
    const _Float16* __restrict__ Wk16, const float* __restrict__ bk,
    const _Float16* __restrict__ Wv16, const float* __restrict__ bv,
    const float* __restrict__ Wo, const float* __restrict__ bo,
    const float* __restrict__ m1_W, const float* __restrict__ m1_b,
    const float* __restrict__ m2_W, const float* __restrict__ m2_b,
    float* __restrict__ emb_out)
{
    __shared__ _Float16 s_kvin[32][DK_];     // 24576 B (rows 20..31 zero)
    __shared__ _Float16 s_K[KNBR_][DQ_];     // 10240 B
    __shared__ _Float16 s_V[KNBR_][DQ_];     // 10240 B
    __shared__ float s_srcfeat[D_];
    __shared__ float s_qtime[D_];
    __shared__ float s_q[DQ_];
    __shared__ float s_attn[DQ_];
    __shared__ float s_out[DQ_];
    __shared__ float s_emb1[D_];
    __shared__ float s_prob[2][KNBR_];
    __shared__ int   s_nbr[KNBR_];
    __shared__ int   s_eidx[KNBR_];
    __shared__ float s_dt[KNBR_];
    __shared__ int   s_node;
    __shared__ float s_t;

    const int m   = blockIdx.x;
    const int tid = threadIdx.x;

    if (tid == 0) {
        int node;
        if (m < BATCH_)        node = src_ids[m];
        else if (m < 2*BATCH_) node = dst_ids[m - BATCH_];
        else                   node = neg_ids[m - 2*BATCH_];
        s_node = node;
        s_t = edge_times[m % BATCH_];
    }
    __syncthreads();
    const int node   = s_node;
    const float tcur = s_t;

    if (tid < D_) {
        s_srcfeat[tid] = memory[(size_t)node*D_ + tid] + node_feats[(size_t)node*D_ + tid];
        s_qtime[tid]   = cosf(time_b[tid]);            // cos(0*w + b)
    }
    if (tid < KNBR_) {
        s_nbr[tid]  = neigh_ids[(size_t)node*KNBR_ + tid];
        s_eidx[tid] = neigh_eidx[(size_t)node*KNBR_ + tid];
        s_dt[tid]   = tcur - neigh_times[(size_t)node*KNBR_ + tid];
    }
    __syncthreads();

    // Build kv_in = [nbr_feat | e_feat | time_enc] as f16 (rows >= 20 zero).
    for (int idx = tid; idx < 32*DK_; idx += 256) {
        int kr = idx / DK_, c = idx % DK_;
        float v = 0.0f;
        if (kr < KNBR_) {
            if (c < D_) {
                int nb = s_nbr[kr];
                v = memory[(size_t)nb*D_ + c] + node_feats[(size_t)nb*D_ + c];
            } else if (c < 2*D_) {
                v = edge_feats[(size_t)s_eidx[kr]*D_ + (c - D_)];
            } else {
                int j = c - 2*D_;
                v = cosf(s_dt[kr]*time_w[j] + time_b[j]);
            }
        }
        s_kvin[kr][c] = (_Float16)v;
    }

    // q = [src_feat | q_time] @ Wq + bq  (f32 matrix-vector; 1 col/thread)
    {
        int n = tid;
        float acc = bq[n];
        for (int c = 0; c < D_; ++c) acc += s_srcfeat[c] * Wq[c*DQ_ + n];
        for (int c = 0; c < D_; ++c) acc += s_qtime[c]   * Wq[(D_ + c)*DQ_ + n];
        s_q[n] = acc;
    }
    __syncthreads();

    // ---- WMMA K/V projection: 64 tiles over 8 waves, 12 k-steps each ----
    const int wave   = tid >> 5;
    const int lane   = tid & 31;
    const int laneHi = (lane >= 16) ? 1 : 0;
    const int lcol   = lane & 15;

    for (int ti = 0; ti < 8; ++ti) {
        int T   = wave*8 + ti;     // 0..63
        int mat = T >> 5;          // 0 -> K (Wk), 1 -> V (Wv)
        int mt  = (T >> 4) & 1;    // M tile
        int nt  = T & 15;          // N tile
        const _Float16* wp   = mat ? Wv16 : Wk16;
        const float*    bias = mat ? bv   : bk;
        v8f acc = {};
        for (int kk = 0; kk < 12; ++kk) {
            v16h a, b;
            const _Float16* ar = &s_kvin[mt*16 + lcol][kk*32 + laneHi*16];
#pragma unroll
            for (int e = 0; e < 8; ++e) { a[e] = ar[e]; a[e+8] = ar[8 + e]; }
            const _Float16* bp = wp + (((kk*16 + nt) << 9) | (lane << 4));
#pragma unroll
            for (int e = 0; e < 16; ++e) b[e] = bp[e];
            acc = __builtin_amdgcn_wmma_f32_16x16x32_f16(
                false, a, false, b, (short)0, acc, false, false);
        }
#pragma unroll
        for (int i = 0; i < 8; ++i) {
            int row = mt*16 + i + laneHi*8;
            int col = nt*16 + lcol;
            if (row < KNBR_) {
                float v = acc[i] + bias[col];
                if (mat) s_V[row][col] = (_Float16)v;
                else     s_K[row][col] = (_Float16)v;
            }
        }
    }
    __syncthreads();

    // scores (2 heads x 20 nbrs), pad mask, softmax
    if (tid < 2*KNBR_) {
        int h = tid / KNBR_, k = tid % KNBR_;
        float acc = 0.0f;
        for (int d = 0; d < D_; ++d) acc += s_q[h*D_ + d] * (float)s_K[k][h*D_ + d];
        acc *= 0.088388347648318447f;   // 1/sqrt(128)
        if (s_nbr[k] == 0) acc = -1.0e9f;
        s_prob[h][k] = acc;
    }
    __syncthreads();
    if (tid < 2) {
        int h = tid;
        float mx = -3.0e38f;
#pragma unroll
        for (int k = 0; k < KNBR_; ++k) mx = fmaxf(mx, s_prob[h][k]);
        float ev[KNBR_]; float sum = 0.0f;
#pragma unroll
        for (int k = 0; k < KNBR_; ++k) { ev[k] = expf(s_prob[h][k] - mx); sum += ev[k]; }
        float inv = 1.0f / sum;
#pragma unroll
        for (int k = 0; k < KNBR_; ++k) s_prob[h][k] = ev[k] * inv;
    }
    __syncthreads();

    // attn = softmax @ V  (col j per thread)
    {
        int j = tid, h = j >> 7;
        float acc = 0.0f;
#pragma unroll
        for (int k = 0; k < KNBR_; ++k) acc += s_prob[h][k] * (float)s_V[k][j];
        s_attn[j] = acc;
    }
    __syncthreads();
    // out = attn @ Wo + bo
    {
        int n = tid;
        float acc = bo[n];
        for (int c = 0; c < DQ_; ++c) acc += s_attn[c] * Wo[c*DQ_ + n];
        s_out[n] = acc;
    }
    __syncthreads();
    // emb1 = relu([out | src_feat] @ m1_W + m1_b)
    if (tid < D_) {
        int dd = tid;
        float acc = m1_b[dd];
        for (int c = 0; c < DQ_; ++c) acc += s_out[c] * m1_W[c*D_ + dd];
        for (int c = 0; c < D_; ++c)  acc += s_srcfeat[c] * m1_W[(DQ_ + c)*D_ + dd];
        s_emb1[dd] = fmaxf(acc, 0.0f);
    }
    __syncthreads();
    // emb = emb1 @ m2_W + m2_b
    if (tid < D_) {
        int dd = tid;
        float acc = m2_b[dd];
        for (int c = 0; c < D_; ++c) acc += s_emb1[c] * m2_W[c*D_ + dd];
        emb_out[(size_t)m*D_ + dd] = acc;
    }
}

// ---------------------------------------------------------------------------
// Affinity head: one block (128 thr) per row r in 0..3999, writes probs.
// ---------------------------------------------------------------------------
__global__ __launch_bounds__(128) void affinity_kernel(
    const float* __restrict__ emb,
    const float* __restrict__ aff1_W, const float* __restrict__ aff1_b,
    const float* __restrict__ aff2_W, const float* __restrict__ aff2_b,
    float* __restrict__ probs)
{
    __shared__ float s_x1[D_], s_x2[D_], s_h[D_];
    int r = blockIdx.x, tid = threadIdx.x;
    int i1 = r % BATCH_;                                         // src_emb
    int i2 = (r < BATCH_) ? (BATCH_ + r) : (2*BATCH_ + (r - BATCH_)); // dst / neg
    s_x1[tid] = emb[(size_t)i1*D_ + tid];
    s_x2[tid] = emb[(size_t)i2*D_ + tid];
    __syncthreads();
    float acc = aff1_b[tid];
    for (int c = 0; c < D_; ++c) acc += s_x1[c] * aff1_W[c*D_ + tid];
    for (int c = 0; c < D_; ++c) acc += s_x2[c] * aff1_W[(D_ + c)*D_ + tid];
    s_h[tid] = fmaxf(acc, 0.0f) * aff2_W[tid];
    __syncthreads();
    if (tid == 0) {
        float s = aff2_b[0];
        for (int c = 0; c < D_; ++c) s += s_h[c];
        probs[r] = sigmoidf_(s);
    }
}

// ---------------------------------------------------------------------------
// Message MLP + GRU: one block (256 thr) per row i in 0..3999 -> h_new.
// ---------------------------------------------------------------------------
__global__ __launch_bounds__(256) void message_gru_kernel(
    const float* __restrict__ memory, const float* __restrict__ edge_feats,
    const float* __restrict__ last_update, const float* __restrict__ edge_times,
    const int* __restrict__ src_ids, const int* __restrict__ dst_ids,
    const int* __restrict__ edge_idxes,
    const float* __restrict__ time_w, const float* __restrict__ time_b,
    const float* __restrict__ msg1_W, const float* __restrict__ msg1_b,
    const float* __restrict__ msg2_W, const float* __restrict__ msg2_b,
    const float* __restrict__ gru_Wih, const float* __restrict__ gru_bih,
    const float* __restrict__ gru_Whh, const float* __restrict__ gru_bhh,
    float* __restrict__ hnew)
{
    __shared__ float s_msg[RAW_];
    __shared__ float s_h1[RAW_/2];
    __shared__ float s_m[DMSG_];
    __shared__ float s_gi[3*D_];
    __shared__ float s_gh[3*D_];
    __shared__ float s_hold[D_];

    int i = blockIdx.x, tid = threadIdx.x;
    int b = i % BATCH_;
    int sn, dn;
    if (i < BATCH_) { sn = src_ids[b]; dn = dst_ids[b]; }
    else            { sn = dst_ids[b]; dn = src_ids[b]; }
    float t  = edge_times[b];
    int  eix = edge_idxes[b];

    for (int c = tid; c < RAW_; c += 256) {
        float v;
        if (c < D_)        v = memory[(size_t)sn*D_ + c];
        else if (c < 2*D_) v = memory[(size_t)dn*D_ + (c - D_)];
        else if (c < 3*D_) v = edge_feats[(size_t)eix*D_ + (c - 2*D_)];
        else { int j = c - 3*D_; v = cosf((t - last_update[sn])*time_w[j] + time_b[j]); }
        s_msg[c] = v;
    }
    if (tid < D_) s_hold[tid] = memory[(size_t)sn*D_ + tid];
    __syncthreads();

    { // h1 = relu(msg @ msg1_W + b)  (512 -> 256)
        int n = tid;
        float acc = msg1_b[n];
        for (int c = 0; c < RAW_; ++c) acc += s_msg[c] * msg1_W[c*(RAW_/2) + n];
        s_h1[n] = fmaxf(acc, 0.0f);
    }
    __syncthreads();
    if (tid < DMSG_) { // m = h1 @ msg2_W + b  (256 -> 100)
        int n = tid;
        float acc = msg2_b[n];
        for (int c = 0; c < RAW_/2; ++c) acc += s_h1[c] * msg2_W[c*DMSG_ + n];
        s_m[n] = acc;
    }
    __syncthreads();
    for (int n = tid; n < 3*D_; n += 256) {
        float a1 = gru_bih[n];
        for (int c = 0; c < DMSG_; ++c) a1 += s_m[c] * gru_Wih[c*(3*D_) + n];
        s_gi[n] = a1;
        float a2 = gru_bhh[n];
        for (int c = 0; c < D_; ++c) a2 += s_hold[c] * gru_Whh[c*(3*D_) + n];
        s_gh[n] = a2;
    }
    __syncthreads();
    if (tid < D_) {
        int dd = tid;
        float r = sigmoidf_(s_gi[dd]        + s_gh[dd]);
        float z = sigmoidf_(s_gi[D_ + dd]   + s_gh[D_ + dd]);
        float g = tanhf(s_gi[2*D_ + dd] + r * s_gh[2*D_ + dd]);
        hnew[(size_t)i*D_ + dd] = (1.0f - z)*g + z*s_hold[dd];
    }
}

// ---------------------------------------------------------------------------
// State copy, last-position dedup, scatter.
// ---------------------------------------------------------------------------
__global__ void copy_state_kernel(const float* __restrict__ memory,
                                  const float* __restrict__ last_update,
                                  float* __restrict__ out_mem, float* __restrict__ out_last) {
    size_t idx = (size_t)blockIdx.x*256 + threadIdx.x;
    const size_t tm = (size_t)N_NODES_*D_;
    if (idx < tm) out_mem[idx] = memory[idx];
    else if (idx < tm + N_NODES_) out_last[idx - tm] = last_update[idx - tm];
}

__global__ void init_lastpos_kernel(int* __restrict__ lastpos) {
    int i = blockIdx.x*256 + threadIdx.x;
    if (i < N_NODES_) lastpos[i] = -1;
}

__global__ void mark_lastpos_kernel(const int* __restrict__ src_ids,
                                    const int* __restrict__ dst_ids,
                                    int* __restrict__ lastpos) {
    int i = blockIdx.x*256 + threadIdx.x;
    if (i < IDROWS_) {
        int id = (i < BATCH_) ? src_ids[i] : dst_ids[i - BATCH_];
        atomicMax(&lastpos[id], i);
    }
}

__global__ __launch_bounds__(128) void scatter_kernel(
    const int* __restrict__ src_ids, const int* __restrict__ dst_ids,
    const float* __restrict__ edge_times, const int* __restrict__ lastpos,
    const float* __restrict__ hnew,
    float* __restrict__ out_mem, float* __restrict__ out_last) {
    int i = blockIdx.x, tid = threadIdx.x;
    int id = (i < BATCH_) ? src_ids[i] : dst_ids[i - BATCH_];
    if (lastpos[id] == i) {
        out_mem[(size_t)id*D_ + tid] = hnew[(size_t)i*D_ + tid];
        if (tid == 0) out_last[id] = edge_times[i % BATCH_];
    }
}

// ---------------------------------------------------------------------------
extern "C" void kernel_launch(void* const* d_in, const int* in_sizes, int n_in,
                              void* d_out, int out_size, void* d_ws, size_t ws_size,
                              hipStream_t stream) {
    (void)in_sizes; (void)n_in; (void)out_size; (void)ws_size;
    // inputs (setup_inputs dict order; params dict flattened in insertion order)
    const float* node_feats  = (const float*)d_in[0];
    const float* edge_feats  = (const float*)d_in[1];
    const float* memory      = (const float*)d_in[2];
    const float* last_update = (const float*)d_in[3];
    const float* edge_times  = (const float*)d_in[4];
    const float* neigh_times = (const float*)d_in[5];
    const int*   src_ids     = (const int*)d_in[6];
    const int*   dst_ids     = (const int*)d_in[7];
    const int*   neg_ids     = (const int*)d_in[8];
    const int*   edge_idxes  = (const int*)d_in[9];
    const int*   neigh_ids   = (const int*)d_in[10];
    const int*   neigh_eidx  = (const int*)d_in[11];
    const float* time_w  = (const float*)d_in[12];
    const float* time_b  = (const float*)d_in[13];
    const float* Wq      = (const float*)d_in[14];
    const float* bq      = (const float*)d_in[15];
    const float* Wk      = (const float*)d_in[16];
    const float* bk      = (const float*)d_in[17];
    const float* Wv      = (const float*)d_in[18];
    const float* bv      = (const float*)d_in[19];
    const float* Wo      = (const float*)d_in[20];
    const float* bo      = (const float*)d_in[21];
    const float* m1_W    = (const float*)d_in[22];
    const float* m1_b    = (const float*)d_in[23];
    const float* m2_W    = (const float*)d_in[24];
    const float* m2_b    = (const float*)d_in[25];
    const float* aff1_W  = (const float*)d_in[26];
    const float* aff1_b  = (const float*)d_in[27];
    const float* aff2_W  = (const float*)d_in[28];
    const float* aff2_b  = (const float*)d_in[29];
    const float* msg1_W  = (const float*)d_in[30];
    const float* msg1_b  = (const float*)d_in[31];
    const float* msg2_W  = (const float*)d_in[32];
    const float* msg2_b  = (const float*)d_in[33];
    const float* gru_Wih = (const float*)d_in[34];
    const float* gru_bih = (const float*)d_in[35];
    const float* gru_Whh = (const float*)d_in[36];
    const float* gru_bhh = (const float*)d_in[37];

    // d_out layout: [probs 4000][new_memory 200000*128][new_last 200000]
    float* probs    = (float*)d_out;
    float* out_mem  = probs + IDROWS_;
    float* out_last = out_mem + (size_t)N_NODES_*D_;

    // workspace layout (bytes, 256B-aligned)
    char* ws = (char*)d_ws;
    _Float16* Wk16 = (_Float16*)(ws + 0);                 // 196608 B
    _Float16* Wv16 = (_Float16*)(ws + 196608);            // 196608 B
    float* emb     = (float*)(ws + 393216);               // 6000*128*4 = 3072000 B
    float* hnew    = (float*)(ws + 3465216);              // 4000*128*4 = 2048000 B
    int*   lastpos = (int*)(ws + 5513216);                // 200000*4   =  800000 B

    // 1) weight repack to f16 WMMA fragments
    repack_kernel<<<(2*98304 + 255)/256, 256, 0, stream>>>(Wk, Wv, Wk16, Wv16);
    // 2) attention + embedding (WMMA core)
    attn_emb_kernel<<<MROWS_, 256, 0, stream>>>(
        node_feats, memory, edge_feats, edge_times, neigh_times,
        src_ids, dst_ids, neg_ids, neigh_ids, neigh_eidx,
        time_w, time_b, Wq, bq, Wk16, bk, Wv16, bv, Wo, bo,
        m1_W, m1_b, m2_W, m2_b, emb);
    // 3) affinity probs
    affinity_kernel<<<IDROWS_, 128, 0, stream>>>(emb, aff1_W, aff1_b, aff2_W, aff2_b, probs);
    // 4) messages + GRU
    message_gru_kernel<<<IDROWS_, 256, 0, stream>>>(
        memory, edge_feats, last_update, edge_times, src_ids, dst_ids, edge_idxes,
        time_w, time_b, msg1_W, msg1_b, msg2_W, msg2_b,
        gru_Wih, gru_bih, gru_Whh, gru_bhh, hnew);
    // 5) copy memory/last_update into outputs
    {
        size_t total = (size_t)N_NODES_*D_ + N_NODES_;
        copy_state_kernel<<<(unsigned)((total + 255)/256), 256, 0, stream>>>(
            memory, last_update, out_mem, out_last);
    }
    // 6) dedup scatter targets (lastpos = max position per id)
    init_lastpos_kernel<<<(N_NODES_ + 255)/256, 256, 0, stream>>>(lastpos);
    mark_lastpos_kernel<<<(IDROWS_ + 255)/256, 256, 0, stream>>>(src_ids, dst_ids, lastpos);
    // 7) scatter h_new / t2 for winning rows
    scatter_kernel<<<IDROWS_, 128, 0, stream>>>(
        src_ids, dst_ids, edge_times, lastpos, hnew, out_mem, out_last);
}
